// MultiHeadAttention_24532853195009
// MI455X (gfx1250) — matrix operand order repaired and checked by reading, not compiled, not run on hardware
//
#include <hip/hip_runtime.h>
#include <hip/hip_bf16.h>

typedef __bf16 bf16;
typedef bf16  v8bf  __attribute__((ext_vector_type(8)));
typedef bf16  v16bf __attribute__((ext_vector_type(16)));
typedef float v4f   __attribute__((ext_vector_type(4)));
typedef float v8f   __attribute__((ext_vector_type(8)));

#define Bsz 4
#define S   2048
#define DM  512
#define H   4
#define DEP 128

__device__ __forceinline__ v8f wmma_bf16(v16bf a, v16bf b, v8f c) {
    return __builtin_amdgcn_wmma_f32_16x16x32_bf16(false, a, false, b, (short)0, c, false, false);
}

// A-operand (16-bit 16x32): lane half h gets K = 8h..8h+7 (elems 0..7) and 8h+16..8h+23 (elems 8..15)
__device__ __forceinline__ v16bf load_afrag_bf16(const bf16* rowptr, int kb) {
    v8bf lo = *(const v8bf*)(rowptr + kb);
    v8bf hi = *(const v8bf*)(rowptr + kb + 16);
    return __builtin_shufflevector(lo, hi, 0,1,2,3,4,5,6,7,8,9,10,11,12,13,14,15);
}

// Same A layout but source is f32 (convert on the fly)
__device__ __forceinline__ v16bf load_afrag_f32(const float* rowptr, int kb) {
    v4f f0 = *(const v4f*)(rowptr + kb);
    v4f f1 = *(const v4f*)(rowptr + kb + 4);
    v4f f2 = *(const v4f*)(rowptr + kb + 16);
    v4f f3 = *(const v4f*)(rowptr + kb + 20);
    v16bf a;
#pragma unroll
    for (int j = 0; j < 4; ++j) {
        a[j]      = (bf16)f0[j];
        a[4 + j]  = (bf16)f1[j];
        a[8 + j]  = (bf16)f2[j];
        a[12 + j] = (bf16)f3[j];
    }
    return a;
}

// B-operand (16-bit 32x16): lane half h gets K = 8h..8h+7 (elems 0..7) and 8h+16..8h+23 (elems 8..15)
__device__ __forceinline__ v16bf load_bfrag_bf16(const bf16* rowptr, int kb) {
    int t = (kb & ~31) + ((kb & 16) >> 1);
    v8bf lo = *(const v8bf*)(rowptr + t);
    v8bf hi = *(const v8bf*)(rowptr + t + 16);
    return __builtin_shufflevector(lo, hi, 0,1,2,3,4,5,6,7,8,9,10,11,12,13,14,15);
}

// ---------------- K0: weight f32 -> bf16 transposed ----------------
__global__ void transpose_w(const float* __restrict__ w0, const float* __restrict__ w1,
                            const float* __restrict__ w2, const float* __restrict__ w3,
                            bf16* __restrict__ out) {
    int mat = blockIdx.y;
    const float* in = (mat == 0) ? w0 : (mat == 1) ? w1 : (mat == 2) ? w2 : w3;
    bf16* o = out + (size_t)mat * DM * DM;
    int idx = blockIdx.x * 256 + threadIdx.x;      // idx = k*512 + n
    int k = idx >> 9, n = idx & 511;
    o[n * DM + k] = (bf16)in[idx];
}

// ---------------- K1: QKV projection GEMM (f32 X * bf16 W^T) ----------------
// mode 0/1: write [b,h,s,d] (Q,K).  mode 2: write [b,h,d,s] (V, depth-major).
__global__ __launch_bounds__(256) void gemm_qkv(const float* __restrict__ X,
                                                const bf16* __restrict__ WT,
                                                const float* __restrict__ bias,
                                                bf16* __restrict__ out, int mode) {
    const int wave = threadIdx.x >> 5;
    const int lane = threadIdx.x & 31;
    const int lmod = lane & 15;
    const int loff = (lane >> 4) * 8;   // A/C lane-half offset
    const int boff = (lane >> 4) * 16;  // B lane-half offset
    const int mtile = blockIdx.x * 8 + wave;   // 0..511
    const int n = blockIdx.y * 16 + lmod;      // output column for this lane
    const float* xrow = X + (size_t)(mtile * 16 + lmod) * DM;
    const bf16*  wrow = WT + (size_t)n * DM;
    v8f c = {};
#pragma unroll 4
    for (int kk = 0; kk < 16; ++kk) {
        v16bf a = load_afrag_f32(xrow, kk * 32 + loff);
        v16bf b = load_bfrag_bf16(wrow, kk * 32 + boff);
        c = wmma_bf16(a, b, c);
    }
    const float bv = bias[n];
    const int h = n >> 7, d = n & 127;
    if (mode < 2) {
#pragma unroll
        for (int i = 0; i < 8; ++i) {
            int m = mtile * 16 + loff + i;
            int bb = m >> 11, s = m & (S - 1);
            out[(((size_t)(bb * H + h) * S + s) << 7) + d] = (bf16)(c[i] + bv);
        }
    } else {
        int m0 = mtile * 16 + loff;
        int bb = m0 >> 11, s0 = m0 & (S - 1);
        v8bf t;
#pragma unroll
        for (int i = 0; i < 8; ++i) t[i] = (bf16)(c[i] + bv);
        *(v8bf*)(out + ((size_t)(bb * H + h) * DEP + d) * S + s0) = t;  // contiguous in s
    }
}

// ---------------- K2: fused scores + softmax + PV ----------------
__global__ __launch_bounds__(256) void attn_kernel(const bf16* __restrict__ qh,
                                                   const bf16* __restrict__ kh,
                                                   const bf16* __restrict__ vt,
                                                   const int*  __restrict__ mask,
                                                   float* __restrict__ attn_out,
                                                   bf16*  __restrict__ ctx) {
    __shared__ float sc[16][S];   // 128 KB score tile (fits 320KB WGP LDS)
    const int bidx = blockIdx.x;
    const int qt = bidx & 127, hh = (bidx >> 7) & 3, bb = bidx >> 9;
    const int q0 = qt * 16;
    const int tid = threadIdx.x, wave = tid >> 5, lane = tid & 31;
    const int lmod = lane & 15, loff = (lane >> 4) * 8, boff = (lane >> 4) * 16;
    const bf16* qbase = qh + (size_t)(bb * H + hh) * S * DEP;
    const bf16* kbase = kh + (size_t)(bb * H + hh) * S * DEP;
    const bf16* vbase = vt + (size_t)(bb * H + hh) * DEP * S;

    // Preload the Q A-fragments for this 16-row tile (depth = 4 K-steps)
    const bf16* qrow = qbase + (size_t)(q0 + lmod) * DEP;
    v16bf aq[4];
#pragma unroll
    for (int kk = 0; kk < 4; ++kk) aq[kk] = load_afrag_bf16(qrow, kk * 32 + loff);

    const float scale = 0.08838834764831845f;  // 1/sqrt(128)
    // --- Phase 1: scores -> LDS (waves own disjoint key tiles) ---
    for (int jt = wave; jt < 128; jt += 8) {
        int k0 = jt * 16;
        const bf16* krow = kbase + (size_t)(k0 + lmod) * DEP;
        v8f c = {};
#pragma unroll
        for (int kk = 0; kk < 4; ++kk)
            c = wmma_bf16(aq[kk], load_bfrag_bf16(krow, kk * 32 + boff), c);
        const int col = k0 + lmod;
        const int* mrow = mask + ((size_t)bb * S + q0) * S + col;
#pragma unroll
        for (int i = 0; i < 8; ++i) {
            int M = i + loff;
            sc[M][col] = c[i] * scale + (float)mrow[(size_t)M * S] * -1e9f;
        }
    }
    __syncthreads();

    // --- Phase 2: softmax over each of the 16 rows; write attn to global ---
    {
        int r = tid >> 4, cb = tid & 15;   // 16 threads per row
        float mx = -3.402823466e38f;
#pragma unroll 8
        for (int i = 0; i < 128; ++i) mx = fmaxf(mx, sc[r][cb + 16 * i]);
        for (int o = 8; o; o >>= 1) mx = fmaxf(mx, __shfl_xor(mx, o, 16));
        float sum = 0.f;
#pragma unroll 8
        for (int i = 0; i < 128; ++i) {
            float e = __expf(sc[r][cb + 16 * i] - mx);
            sc[r][cb + 16 * i] = e;
            sum += e;
        }
        for (int o = 8; o; o >>= 1) sum += __shfl_xor(sum, o, 16);
        float inv = 1.0f / sum;
        float* arow = attn_out + ((size_t)(bb * H + hh) * S + q0 + r) * S;
#pragma unroll 8
        for (int i = 0; i < 128; ++i) {
            int cc = cb + 16 * i;
            float p = sc[r][cc] * inv;
            sc[r][cc] = p;
            arow[cc] = p;
        }
    }
    __syncthreads();

    // --- Phase 3: PV. wave -> 16-wide depth column; K-loop over 2048 keys ---
    const int dcol = wave;
    const bf16* vrow = vbase + (size_t)(dcol * 16 + lmod) * S;  // depth-major V
    v8f c = {};
    for (int kk = 0; kk < 64; ++kk) {
        int kbA = kk * 32 + loff;
        v4f f0 = *(const v4f*)&sc[lmod][kbA];
        v4f f1 = *(const v4f*)&sc[lmod][kbA + 4];
        v4f f2 = *(const v4f*)&sc[lmod][kbA + 16];
        v4f f3 = *(const v4f*)&sc[lmod][kbA + 20];
        v16bf a;
#pragma unroll
        for (int j = 0; j < 4; ++j) {
            a[j] = (bf16)f0[j]; a[4 + j] = (bf16)f1[j];
            a[8 + j] = (bf16)f2[j]; a[12 + j] = (bf16)f3[j];
        }
        v16bf b = load_bfrag_bf16(vrow, kk * 32 + boff);
        c = wmma_bf16(a, b, c);
    }
#pragma unroll
    for (int i = 0; i < 8; ++i) {
        int s = q0 + loff + i;
        ctx[((size_t)bb * S + s) * DM + hh * DEP + dcol * 16 + lmod] = (bf16)c[i];
    }
}

// ---------------- K3: output projection (bf16 ctx * bf16 proj^T -> f32) ----------------
__global__ __launch_bounds__(256) void gemm_out(const bf16* __restrict__ ctx,
                                                const bf16* __restrict__ WT,
                                                const float* __restrict__ bias,
                                                float* __restrict__ out) {
    const int wave = threadIdx.x >> 5;
    const int lane = threadIdx.x & 31;
    const int lmod = lane & 15;
    const int loff = (lane >> 4) * 8;
    const int boff = (lane >> 4) * 16;
    const int mtile = blockIdx.x * 8 + wave;
    const int n = blockIdx.y * 16 + lmod;
    const bf16* arow = ctx + (size_t)(mtile * 16 + lmod) * DM;
    const bf16* wrow = WT + (size_t)n * DM;
    v8f c = {};
#pragma unroll 4
    for (int kk = 0; kk < 16; ++kk) {
        v16bf a = load_afrag_bf16(arow, kk * 32 + loff);
        v16bf b = load_bfrag_bf16(wrow, kk * 32 + boff);
        c = wmma_bf16(a, b, c);
    }
    const float bv = bias[n];
#pragma unroll
    for (int i = 0; i < 8; ++i)
        out[(size_t)(mtile * 16 + loff + i) * DM + n] = c[i] + bv;
}

extern "C" void kernel_launch(void* const* d_in, const int* in_sizes, int n_in,
                              void* d_out, int out_size, void* d_ws, size_t ws_size,
                              hipStream_t stream) {
    const float* v_in   = (const float*)d_in[0];
    const float* k_in   = (const float*)d_in[1];
    const float* q_in   = (const float*)d_in[2];
    const int*   mask   = (const int*)  d_in[3];
    const float* wq_w   = (const float*)d_in[4];
    const float* wq_b   = (const float*)d_in[5];
    const float* wk_w   = (const float*)d_in[6];
    const float* wk_b   = (const float*)d_in[7];
    const float* wv_w   = (const float*)d_in[8];
    const float* wv_b   = (const float*)d_in[9];
    const float* proj_w = (const float*)d_in[10];
    const float* proj_b = (const float*)d_in[11];

    float* out  = (float*)d_out;                       // [B,S,512]
    float* attn = out + (size_t)Bsz * S * DM;          // [B,H,S,S]

    const size_t HEAD_ELEMS = (size_t)Bsz * H * S * DEP;  // 4M bf16 = 8MB
    char* ws = (char*)d_ws;
    bf16* qh  = (bf16*)(ws);
    bf16* kh  = (bf16*)(ws + 8u  * 1024 * 1024);
    bf16* vt  = (bf16*)(ws + 16u * 1024 * 1024);
    bf16* ctx = (bf16*)(ws + 24u * 1024 * 1024);
    bf16* wt  = (bf16*)(ws + 32u * 1024 * 1024);       // 4x [512][512] bf16 transposed
    (void)HEAD_ELEMS; (void)in_sizes; (void)n_in; (void)out_size; (void)ws_size;

    // K0: weights -> bf16 transposed
    transpose_w<<<dim3((DM * DM) / 256, 4), 256, 0, stream>>>(wq_w, wk_w, wv_w, proj_w, wt);
    // K1: Q, K (token-major) and V (depth-major) projections
    gemm_qkv<<<dim3(64, 32), 256, 0, stream>>>(q_in, wt,               wq_b, qh, 0);
    gemm_qkv<<<dim3(64, 32), 256, 0, stream>>>(k_in, wt + 262144,      wk_b, kh, 1);
    gemm_qkv<<<dim3(64, 32), 256, 0, stream>>>(v_in, wt + 2 * 262144,  wv_b, vt, 2);
    // K2: fused attention (scores + softmax + PV), writes attn once
    attn_kernel<<<dim3(Bsz * H * (S / 16)), 256, 0, stream>>>(qh, kh, vt, mask, attn, ctx);
    // K3: output projection
    gemm_out<<<dim3(64, 32), 256, 0, stream>>>(ctx, wt + 3 * 262144, proj_b, out);
}
